// ContextTransformer_63505386439229
// MI455X (gfx1250) — matrix operand order repaired and checked by reading, not compiled, not run on hardware
//
#include <hip/hip_runtime.h>

// ---------------- model dims ----------------
#define B_    64
#define T_    256
#define MF    222      // motion features
#define DPH   16       // phase dim
#define NEXP  8        // experts
#define DMOD  512
#define NHEAD 8
#define DHEAD 64
#define NLAY  6
#define PFF   2048
#define NTOK  (B_*T_)      // 16384
#define NKEY  11           // attendable columns: 0..9 and 255
#define KPAD  12
#define KROWS (B_*KPAD)    // 768

typedef __attribute__((ext_vector_type(16))) _Float16 v16h;
typedef __attribute__((ext_vector_type(8)))  float    v8f;

enum { ACT_NONE = 0, ACT_PRELU = 1, ACT_RELU = 2 };

// ---------------- WMMA helpers ----------------

__device__ __forceinline__ v8f wmma16(v16h a, v16h b, v8f c) {
  // D = A(16x32,f16) * B(32x16,f16) + C(16x16,f32)
  return __builtin_amdgcn_wmma_f32_16x16x32_f16(
      false, a, false, b, (short)0, c, false, false);
}

union FragH { v16h v; unsigned u[8]; };
union PackH { _Float16 h[2]; unsigned u; };

// LDS tiles: 128 rows x 32 halves, stored packed as 17 dwords/row (pitch 17
// dwords = 17 banks -> conflict-free strided access across 16 lanes).
#define TPITCH 17
#define TSZ    (128 * TPITCH)

// ---- global -> register tile fetch (branch-free fast path on full tiles) ----

__device__ __forceinline__ void fetch_a(float (&f)[4][4], const float* __restrict__ A,
                                        int lda, int mBase, int k0, int M, int K,
                                        int tid) {
  int tr = tid >> 3;            // 0..31
  int tc = (tid & 7) * 4;       // 0,4,...,28
  if (mBase + 128 <= M && k0 + 32 <= K) {     // uniform scalar branch
#pragma unroll
    for (int p = 0; p < 4; ++p) {
      const float* ap = A + (size_t)(mBase + tr + p * 32) * lda + (k0 + tc);
#pragma unroll
      for (int i = 0; i < 4; ++i) f[p][i] = ap[i];
    }
  } else {
#pragma unroll
    for (int p = 0; p < 4; ++p) {
      int m = mBase + tr + p * 32;
#pragma unroll
      for (int i = 0; i < 4; ++i) {
        int k = k0 + tc + i;
        f[p][i] = (m < M && k < K) ? A[(size_t)m * lda + k] : 0.f;
      }
    }
  }
}

__device__ __forceinline__ void store_a(unsigned* As, const float (&f)[4][4], int tid) {
  int tr = tid >> 3;
  int tc = (tid & 7) * 4;
#pragma unroll
  for (int p = 0; p < 4; ++p) {
    PackH p0, p1;
    p0.h[0] = (_Float16)f[p][0]; p0.h[1] = (_Float16)f[p][1];
    p1.h[0] = (_Float16)f[p][2]; p1.h[1] = (_Float16)f[p][3];
    int row = tr + p * 32;
    As[row * TPITCH + (tc >> 1)]     = p0.u;
    As[row * TPITCH + (tc >> 1) + 1] = p1.u;
  }
}

// W is row-major [K x N]; tile is stored transposed in LDS as [n][k].
__device__ __forceinline__ void fetch_b(float (&f)[8][2], const float* __restrict__ W,
                                        int ldw, int nBase, int k0, int N, int K,
                                        int tid) {
  int n  = tid & 127;
  int kp = tid >> 7;            // 0..1
  int gn = nBase + n;
  if (nBase + 128 <= N && k0 + 32 <= K) {     // uniform scalar branch
#pragma unroll
    for (int it = 0; it < 8; ++it) {
      int k = k0 + (kp + it * 2) * 2;
      const float* wp = W + (size_t)k * ldw + gn;
      f[it][0] = wp[0];
      f[it][1] = wp[ldw];
    }
  } else {
#pragma unroll
    for (int it = 0; it < 8; ++it) {
      int k = k0 + (kp + it * 2) * 2;
      f[it][0] = (gn < N && k     < K) ? W[(size_t)k       * ldw + gn] : 0.f;
      f[it][1] = (gn < N && (k+1) < K) ? W[(size_t)(k + 1) * ldw + gn] : 0.f;
    }
  }
}

__device__ __forceinline__ void store_b(unsigned* Bs, const float (&f)[8][2], int tid) {
  int n  = tid & 127;
  int kp = tid >> 7;
#pragma unroll
  for (int it = 0; it < 8; ++it) {
    int kk = (kp + it * 2) * 2;
    PackH pk; pk.h[0] = (_Float16)f[it][0]; pk.h[1] = (_Float16)f[it][1];
    Bs[n * TPITCH + (kk >> 1)] = pk.u;
  }
}

// A fragment (16x32): lane -> M=lane&15, regs r: K = (lane>>4)*16 + 2r
__device__ __forceinline__ v16h load_a_frag(const unsigned* As, int rowBase, int lane) {
  int hl = lane & 15, hi = lane >> 4;
  FragH f;
  const unsigned* p = As + (rowBase + hl) * TPITCH;
#pragma unroll
  for (int r = 0; r < 8; ++r) {
    int ku = hi * 8 + r;
    f.u[r] = p[ku];
  }
  return f.v;
}

// B fragment (32x16): lane -> N=lane&15, regs r: K = (lane>>4)*16 + 2r
__device__ __forceinline__ v16h load_b_frag(const unsigned* Bs, int nBase, int lane) {
  int hl = lane & 15, hi = lane >> 4;
  FragH f;
  const unsigned* p = Bs + (nBase + hl) * TPITCH + hi * 8;
#pragma unroll
  for (int r = 0; r < 8; ++r) f.u[r] = p[r];
  return f.v;
}

// all fragments up front so ds waits can be staggered against the WMMA pipe
__device__ __forceinline__ void compute_step(const unsigned* As, const unsigned* Bs,
                                             v8f (&acc)[2][4], int wm, int wn,
                                             int lane) {
  v16h a0 = load_a_frag(As, wm * 32,      lane);
  v16h a1 = load_a_frag(As, wm * 32 + 16, lane);
  v16h b0 = load_b_frag(Bs, wn * 64 +  0, lane);
  v16h b1 = load_b_frag(Bs, wn * 64 + 16, lane);
  v16h b2 = load_b_frag(Bs, wn * 64 + 32, lane);
  v16h b3 = load_b_frag(Bs, wn * 64 + 48, lane);
  acc[0][0] = wmma16(a0, b0, acc[0][0]);
  acc[1][0] = wmma16(a1, b0, acc[1][0]);
  acc[0][1] = wmma16(a0, b1, acc[0][1]);
  acc[1][1] = wmma16(a1, b1, acc[1][1]);
  acc[0][2] = wmma16(a0, b2, acc[0][2]);
  acc[1][2] = wmma16(a1, b2, acc[1][2]);
  acc[0][3] = wmma16(a0, b3, acc[0][3]);
  acc[1][3] = wmma16(a1, b3, acc[1][3]);
}

// ---------------- generic WMMA GEMM (double-buffered LDS) ----------------
// C[M,N] = act( A[M,K] @ W[K,N] + bias (+ residual) )
__global__ __launch_bounds__(256) void gemm_wmma(
    const float* __restrict__ A, int lda,
    const float* __restrict__ W, int ldw,
    const float* __restrict__ bias,
    float* __restrict__ C, int ldc,
    const float* __restrict__ residual,
    int M, int N, int K, int act, const float* __restrict__ actp) {
  __shared__ unsigned As[2][TSZ];
  __shared__ unsigned Bs[2][TSZ];
  int tid = threadIdx.x;
  int lane = tid & 31, wave = tid >> 5;
  int wm = wave & 3, wn = wave >> 2;   // 4x2 wave grid: 32 rows x 64 cols each
  int mBase = blockIdx.y * 128, nBase = blockIdx.x * 128;

  v8f acc[2][4];
  v8f zero = {0.f, 0.f, 0.f, 0.f, 0.f, 0.f, 0.f, 0.f};
#pragma unroll
  for (int mi = 0; mi < 2; ++mi)
#pragma unroll
    for (int ni = 0; ni < 4; ++ni) acc[mi][ni] = zero;

  float ap = (act == ACT_PRELU) ? actp[0] : 0.f;

  float af[4][4];
  float bf[8][2];
  fetch_a(af, A, lda, mBase, 0, M, K, tid);
  fetch_b(bf, W, ldw, nBase, 0, N, K, tid);
  store_a(As[0], af, tid);
  store_b(Bs[0], bf, tid);
  __syncthreads();

  int cur = 0;
  for (int k0 = 0; k0 < K; k0 += 32) {
    bool hasNext = (k0 + 32) < K;
    if (hasNext) {                      // prefetch next tile into registers
      fetch_a(af, A, lda, mBase, k0 + 32, M, K, tid);
      fetch_b(bf, W, ldw, nBase, k0 + 32, N, K, tid);
    }
    compute_step(As[cur], Bs[cur], acc, wm, wn, lane);
    if (hasNext) {
      store_a(As[cur ^ 1], af, tid);
      store_b(Bs[cur ^ 1], bf, tid);
      __syncthreads();
      cur ^= 1;
    }
  }

  int hl = lane & 15, hi = lane >> 4;
#pragma unroll
  for (int mi = 0; mi < 2; ++mi)
#pragma unroll
    for (int ni = 0; ni < 4; ++ni) {
      int n  = nBase + wn * 64 + ni * 16 + hl;
      int m0 = mBase + wm * 32 + mi * 16 + hi * 8;
      if (n < N) {
        float bv = bias ? bias[n] : 0.f;
#pragma unroll
        for (int c = 0; c < 8; ++c) {
          int m = m0 + c;
          if (m < M) {
            float v = acc[mi][ni][c] + bv;
            if (residual) v += residual[(size_t)m * ldc + n];
            if (act == ACT_PRELU) v = (v >= 0.f) ? v : ap * v;
            else if (act == ACT_RELU) v = fmaxf(v, 0.f);
            C[(size_t)m * ldc + n] = v;
          }
        }
      }
    }
}

// ---------------- fused soft-MoE GEMM ----------------
// C[M,N] = act( sum_e gw[m,e] * (A[M,K] @ W_e[K,N] + b_e[N]) )
__global__ __launch_bounds__(256) void moe_gemm_wmma(
    const float* __restrict__ A,
    const float* __restrict__ Wall,   // [E,K,N]
    const float* __restrict__ ball,   // [E,N]
    const float* __restrict__ gw,     // [M,8]
    float* __restrict__ C,
    int M, int N, int K, int act, const float* __restrict__ actp) {
  __shared__ unsigned As[2][TSZ];
  __shared__ unsigned Bs[2][TSZ];
  __shared__ float    Gs[128 * 8];
  int tid = threadIdx.x;
  int lane = tid & 31, wave = tid >> 5;
  int wm = wave & 3, wn = wave >> 2;
  int mBase = blockIdx.y * 128, nBase = blockIdx.x * 128;
  int hl = lane & 15, hi = lane >> 4;

  for (int i = tid; i < 128 * 8; i += 256) {
    int row = i >> 3, e = i & 7;
    int m = mBase + row;
    Gs[i] = (m < M) ? gw[(size_t)m * 8 + e] : 0.f;
  }

  v8f zero = {0.f, 0.f, 0.f, 0.f, 0.f, 0.f, 0.f, 0.f};
  v8f accT[2][4];
#pragma unroll
  for (int mi = 0; mi < 2; ++mi)
#pragma unroll
    for (int ni = 0; ni < 4; ++ni) accT[mi][ni] = zero;
  float ap = (act == ACT_PRELU) ? actp[0] : 0.f;

  float af[4][4];
  float bf[8][2];

  for (int e = 0; e < NEXP; ++e) {
    const float* W    = Wall + (size_t)e * K * N;
    const float* bias = ball + (size_t)e * N;
    v8f accE[2][4];
#pragma unroll
    for (int mi = 0; mi < 2; ++mi)
#pragma unroll
      for (int ni = 0; ni < 4; ++ni) accE[mi][ni] = zero;

    fetch_a(af, A, K, mBase, 0, M, K, tid);
    fetch_b(bf, W, N, nBase, 0, N, K, tid);
    store_a(As[0], af, tid);
    store_b(Bs[0], bf, tid);
    __syncthreads();

    int cur = 0;
    for (int k0 = 0; k0 < K; k0 += 32) {
      bool hasNext = (k0 + 32) < K;
      if (hasNext) {
        fetch_a(af, A, K, mBase, k0 + 32, M, K, tid);
        fetch_b(bf, W, N, nBase, k0 + 32, N, K, tid);
      }
      compute_step(As[cur], Bs[cur], accE, wm, wn, lane);
      if (hasNext) {
        store_a(As[cur ^ 1], af, tid);
        store_b(Bs[cur ^ 1], bf, tid);
        __syncthreads();
        cur ^= 1;
      }
    }
    __syncthreads();   // buffers recycled by next expert's prologue

#pragma unroll
    for (int mi = 0; mi < 2; ++mi)
#pragma unroll
      for (int ni = 0; ni < 4; ++ni) {
        int n = nBase + wn * 64 + ni * 16 + hl;
        float bv = (n < N) ? bias[n] : 0.f;
#pragma unroll
        for (int c = 0; c < 8; ++c) {
          int mloc = wm * 32 + mi * 16 + hi * 8 + c;
          float g = Gs[mloc * 8 + e];
          accT[mi][ni][c] += g * (accE[mi][ni][c] + bv);
        }
      }
  }

#pragma unroll
  for (int mi = 0; mi < 2; ++mi)
#pragma unroll
    for (int ni = 0; ni < 4; ++ni) {
      int n  = nBase + wn * 64 + ni * 16 + hl;
      int m0 = mBase + wm * 32 + mi * 16 + hi * 8;
      if (n < N) {
#pragma unroll
        for (int c = 0; c < 8; ++c) {
          int m = m0 + c;
          if (m < M) {
            float v = accT[mi][ni][c];
            if (act == ACT_PRELU) v = (v >= 0.f) ? v : ap * v;
            C[(size_t)m * N + n] = v;
          }
        }
      }
    }
}

// ---------------- small VALU kernels ----------------

// assemble encoder input [NTOK, 239] = [motion*dm, dm, phase*dm]
__global__ void build_x0_kernel(const float* __restrict__ motion,
                                const float* __restrict__ phase,
                                float* __restrict__ x0, int total) {
  for (int idx = blockIdx.x * blockDim.x + threadIdx.x; idx < total;
       idx += gridDim.x * blockDim.x) {
    int c = idx % 239;
    int bt = idx / 239;
    int t = bt & (T_ - 1);
    float dm = (t < 10 || t == T_ - 1) ? 1.f : 0.f;
    float v;
    if (c < MF)        v = motion[(size_t)bt * MF + c] * dm;
    else if (c == MF)  v = dm;
    else               v = phase[(size_t)bt * DPH + (c - MF - 1)] * dm;
    x0[idx] = v;
  }
}

// kf1: kp = (t-9, 255-t) -> prelu(kp @ W[2,512] + b)
__global__ void kf1_kernel(const float* __restrict__ W, const float* __restrict__ b,
                           const float* __restrict__ a, float* __restrict__ out) {
  int idx = blockIdx.x * blockDim.x + threadIdx.x;
  if (idx >= T_ * DMOD) return;
  int t = idx >> 9, d = idx & 511;
  float k0 = (float)t - 9.f, k1 = 255.f - (float)t;
  float v = k0 * W[d] + k1 * W[DMOD + d] + b[d];
  float av = a[0];
  out[idx] = (v >= 0.f) ? v : av * v;
}

// rp1: r = row-255 -> prelu(r * W[1,64] + b)
__global__ void rp1_kernel(const float* __restrict__ W, const float* __restrict__ b,
                           const float* __restrict__ a, float* __restrict__ out) {
  int idx = blockIdx.x * blockDim.x + threadIdx.x;
  if (idx >= (2 * T_ - 1) * DHEAD) return;
  int r = idx >> 6, d = idx & 63;
  float rv = (float)r - 255.f;
  float v = rv * W[d] + b[d];
  float av = a[0];
  out[idx] = (v >= 0.f) ? v : av * v;
}

// x[bt,:] += kpe[t,:]
__global__ void add_kpe_kernel(float* __restrict__ x, const float* __restrict__ kpe,
                               int total) {
  for (int idx = blockIdx.x * blockDim.x + threadIdx.x; idx < total;
       idx += gridDim.x * blockDim.x) {
    int d = idx & 511;
    int t = (idx >> 9) & (T_ - 1);
    x[idx] += kpe[t * DMOD + d];
  }
}

// LayerNorm over 512 features; one wave32 per token
__global__ __launch_bounds__(256) void ln_kernel(const float* __restrict__ x,
                                                 const float* __restrict__ g,
                                                 const float* __restrict__ b,
                                                 float* __restrict__ out, int ntok) {
  int wave = threadIdx.x >> 5, lane = threadIdx.x & 31;
  int tok = blockIdx.x * 8 + wave;
  if (tok >= ntok) return;
  const float* xp = x + (size_t)tok * DMOD;
  float s = 0.f, s2 = 0.f;
  for (int j = lane; j < DMOD; j += 32) {
    float v = xp[j];
    s += v; s2 += v * v;
  }
#pragma unroll
  for (int o = 16; o > 0; o >>= 1) {
    s  += __shfl_xor(s,  o, 32);
    s2 += __shfl_xor(s2, o, 32);
  }
  float mu  = s  * (1.f / DMOD);
  float var = s2 * (1.f / DMOD) - mu * mu;
  float r = rsqrtf(var + 1e-5f);
  float* op = out + (size_t)tok * DMOD;
  for (int j = lane; j < DMOD; j += 32)
    op[j] = (xp[j] - mu) * r * g[j] + b[j];
}

// gather LN output at the 11 keyframe columns: gk[b*12+jj, :] (jj==11 -> 0)
__global__ void gather_kernel(const float* __restrict__ h, float* __restrict__ gk) {
  int idx = blockIdx.x * blockDim.x + threadIdx.x;
  if (idx >= KROWS * DMOD) return;
  int d = idx & 511, row = idx >> 9;
  int b = row / KPAD, jj = row % KPAD;
  float v = 0.f;
  if (jj < NKEY) {
    int col = (jj < 10) ? jj : (T_ - 1);
    v = h[((size_t)(b * T_ + col)) * DMOD + d];
  }
  gk[idx] = v;
}

// sparse keyframe attention: 11 attendable columns, rel-pos bias, softmax, ctx
__global__ void attn_kernel(const float* __restrict__ q,
                            const float* __restrict__ ks,
                            const float* __restrict__ vs,
                            const float* __restrict__ rel,
                            float* __restrict__ ctx) {
  int idx = blockIdx.x * blockDim.x + threadIdx.x;
  if (idx >= B_ * T_ * NHEAD) return;
  int h = idx & 7;
  int i = (idx >> 3) & (T_ - 1);
  int b = idx >> 11;
  const float* qp = q + (((size_t)(b * T_ + i)) * NHEAD + h) * DHEAD;

  float s[NKEY];
  float mx = -1e30f;
#pragma unroll
  for (int j = 0; j < NKEY; ++j) {
    int col = (j < 10) ? j : (T_ - 1);
    const float* kp = ks + (((size_t)(b * KPAD + j)) * NHEAD + h) * DHEAD;
    const float* rp = rel + (size_t)(col - i + (T_ - 1)) * DHEAD;
    float ac = 0.f, bd = 0.f;
    for (int d = 0; d < DHEAD; ++d) {
      float qd = qp[d];
      ac += qd * kp[d];
      bd += qd * rp[d];
    }
    s[j] = (ac + bd) * 0.125f;
    mx = fmaxf(mx, s[j]);
  }
  float den = 0.f;
#pragma unroll
  for (int j = 0; j < NKEY; ++j) { s[j] = __expf(s[j] - mx); den += s[j]; }
  float inv = 1.f / den;
  float* op = ctx + (((size_t)(b * T_ + i)) * NHEAD + h) * DHEAD;
  for (int d0 = 0; d0 < DHEAD; d0 += 16) {
    float o[16];
#pragma unroll
    for (int dd = 0; dd < 16; ++dd) o[dd] = 0.f;
#pragma unroll
    for (int j = 0; j < NKEY; ++j) {
      float a = s[j] * inv;
      const float* vp = vs + (((size_t)(b * KPAD + j)) * NHEAD + h) * DHEAD + d0;
      for (int dd = 0; dd < 16; ++dd) o[dd] += a * vp[dd];
    }
    for (int dd = 0; dd < 16; ++dd) op[d0 + dd] = o[dd];
  }
}

// softmax over 8 experts per token
__global__ void softmax8_kernel(const float* __restrict__ logits,
                                float* __restrict__ gw, int ntok) {
  int t = blockIdx.x * blockDim.x + threadIdx.x;
  if (t >= ntok) return;
  float v[NEXP], mx = -1e30f;
#pragma unroll
  for (int e = 0; e < NEXP; ++e) { v[e] = logits[(size_t)t * NEXP + e]; mx = fmaxf(mx, v[e]); }
  float den = 0.f;
#pragma unroll
  for (int e = 0; e < NEXP; ++e) { v[e] = __expf(v[e] - mx); den += v[e]; }
  float inv = 1.f / den;
#pragma unroll
  for (int e = 0; e < NEXP; ++e) gw[(size_t)t * NEXP + e] = v[e] * inv;
}

// ---------------- host orchestration ----------------

static inline void gemm(hipStream_t s, const float* A, int lda, const float* W, int ldw,
                        const float* bias, float* C, int ldc, const float* res,
                        int M, int N, int K, int act, const float* actp) {
  dim3 g((N + 127) / 128, (M + 127) / 128);
  gemm_wmma<<<g, 256, 0, s>>>(A, lda, W, ldw, bias, C, ldc, res, M, N, K, act, actp);
}

static inline void moe(hipStream_t s, const float* A, const float* Wall, const float* ball,
                       const float* gw, float* C, int M, int N, int K,
                       int act, const float* actp) {
  dim3 g((N + 127) / 128, (M + 127) / 128);
  moe_gemm_wmma<<<g, 256, 0, s>>>(A, Wall, ball, gw, C, M, N, K, act, actp);
}

extern "C" void kernel_launch(void* const* d_in, const int* in_sizes, int n_in,
                              void* d_out, int out_size, void* d_ws, size_t ws_size,
                              hipStream_t stream) {
  (void)in_sizes; (void)n_in; (void)out_size; (void)ws_size;
  const float* motion = (const float*)d_in[0];
  const float* phase  = (const float*)d_in[1];
  auto F = [&](int i) -> const float* { return (const float*)d_in[i]; };

  // params in JAX pytree (sorted-dict) flatten order
  const float* enc1W = F(2);  const float* enc1b = F(3);  const float* enc1a = F(4);
  const float* enc2W = F(5);  const float* enc2b = F(6);  const float* enc2a = F(7);
  const float* g1W = F(8);    const float* g1b = F(9);    const float* g1a = F(10);
  const float* g2W = F(11);   const float* g2b = F(12);   const float* g2a = F(13);
  const float* g3W = F(14);   const float* g3b = F(15);
  const float* kf1W = F(16);  const float* kf1b = F(17);
  const float* kf2W = F(18);  const float* kf2b = F(19);  const float* kfa = F(20);
  const int LBASE = 21, LSTR = 16;  // k.W,k.b,ln1_b,ln1_g,ln2_b,ln2_g,o.W,o.b,q.W,q.b,v.W,v.b,w1.W,w1.b,w2.W,w2.b
  const float* lnfb = F(117); const float* lnfg = F(118);
  const float* moe0W = F(119); const float* moe0b = F(120);
  const float* moe1W = F(121); const float* moe1b = F(122);
  const float* moe2W = F(123); const float* moe2b = F(124);
  const float* moea1 = F(125); const float* moea2 = F(126);
  const float* ph1W = F(127); const float* ph1b = F(128);
  const float* ph2W = F(129); const float* ph2b = F(130); const float* pha = F(131);
  const float* rp1W = F(132); const float* rp1b = F(133);
  const float* rp2W = F(134); const float* rp2b = F(135); const float* rpa = F(136);

  // workspace layout (floats)
  float* ws   = (float*)d_ws;
  float* X    = ws;                       // [NTOK,512]
  float* H    = ws + 8388608;             // [NTOK,512]
  float* Q    = ws + 16777216;            // [NTOK,512]
  float* CTX  = ws + 25165824;            // [NTOK,512] (also 4096x2048 FFN chunk)
  float* GK   = ws + 33554432;            // [768,512] gathered keyframe LN rows
  float* Kb   = GK + 393216;              // [768,512]
  float* Vb   = Kb + 393216;              // [768,512]
  float* KPE  = Vb + 393216;              // [256,512]
  float* KPM  = KPE + 131072;             // [256,512]
  float* REL  = KPM + 131072;             // [511,64]
  float* RELM = REL + 32768;              // [511,64]
  float* PH   = RELM + 32768;             // [NTOK,16]
  float* GL   = PH + 262144;              // [NTOK,8]
  float* GW   = GL + 131072;              // [NTOK,8]
  float* X0   = CTX;                      // [NTOK,239] (CTX unused this early)
  float* OUT  = (float*)d_out;            // [NTOK,222]

  // ---- encoder ----
  build_x0_kernel<<<4096, 256, 0, stream>>>(motion, phase, X0, NTOK * 239);
  gemm(stream, X0, 239, enc1W, DMOD, enc1b, H, DMOD, nullptr,
       NTOK, DMOD, 239, ACT_PRELU, enc1a);
  gemm(stream, H, DMOD, enc2W, DMOD, enc2b, X, DMOD, nullptr,
       NTOK, DMOD, DMOD, ACT_PRELU, enc2a);

  // ---- keyframe position encoding & relative position embedding ----
  kf1_kernel<<<(T_ * DMOD) / 256, 256, 0, stream>>>(kf1W, kf1b, kfa, KPM);
  gemm(stream, KPM, DMOD, kf2W, DMOD, kf2b, KPE, DMOD, nullptr,
       T_, DMOD, DMOD, ACT_NONE, nullptr);
  rp1_kernel<<<128, 256, 0, stream>>>(rp1W, rp1b, rpa, RELM);
  gemm(stream, RELM, DHEAD, rp2W, DHEAD, rp2b, REL, DHEAD, nullptr,
       2 * T_ - 1, DHEAD, DHEAD, ACT_NONE, nullptr);
  add_kpe_kernel<<<8192, 256, 0, stream>>>(X, KPE, NTOK * DMOD);

  // ---- transformer layers ----
  for (int l = 0; l < NLAY; ++l) {
    int pb = LBASE + l * LSTR;
    const float* kW = F(pb + 0);  const float* kb = F(pb + 1);
    const float* l1b = F(pb + 2); const float* l1g = F(pb + 3);
    const float* l2b = F(pb + 4); const float* l2g = F(pb + 5);
    const float* oW = F(pb + 6);  const float* ob = F(pb + 7);
    const float* qW = F(pb + 8);  const float* qb = F(pb + 9);
    const float* vW = F(pb + 10); const float* vb = F(pb + 11);
    const float* w1W = F(pb + 12); const float* w1b = F(pb + 13);
    const float* w2W = F(pb + 14); const float* w2b = F(pb + 15);

    ln_kernel<<<NTOK / 8, 256, 0, stream>>>(X, l1g, l1b, H, NTOK);
    gemm(stream, H, DMOD, qW, DMOD, qb, Q, DMOD, nullptr,
         NTOK, DMOD, DMOD, ACT_NONE, nullptr);
    gather_kernel<<<(KROWS * DMOD) / 256, 256, 0, stream>>>(H, GK);
    gemm(stream, GK, DMOD, kW, DMOD, kb, Kb, DMOD, nullptr,
         KROWS, DMOD, DMOD, ACT_NONE, nullptr);
    gemm(stream, GK, DMOD, vW, DMOD, vb, Vb, DMOD, nullptr,
         KROWS, DMOD, DMOD, ACT_NONE, nullptr);
    attn_kernel<<<(B_ * T_ * NHEAD) / 256, 256, 0, stream>>>(Q, Kb, Vb, REL, CTX);
    gemm(stream, CTX, DMOD, oW, DMOD, ob, X, DMOD, /*res=*/X,
         NTOK, DMOD, DMOD, ACT_NONE, nullptr);

    ln_kernel<<<NTOK / 8, 256, 0, stream>>>(X, l2g, l2b, H, NTOK);
    // FFN chunked x4 so the 2048-wide mid fits in the CTX region
    for (int c = 0; c < 4; ++c) {
      const float* Ain = H + (size_t)c * 4096 * DMOD;
      float* Xc = X + (size_t)c * 4096 * DMOD;
      gemm(stream, Ain, DMOD, w1W, PFF, w1b, CTX, PFF, nullptr,
           4096, PFF, DMOD, ACT_RELU, nullptr);
      gemm(stream, CTX, PFF, w2W, DMOD, w2b, Xc, DMOD, /*res=*/Xc,
           4096, DMOD, PFF, ACT_NONE, nullptr);
    }
  }

  // ---- final LN -> zL in H ----
  ln_kernel<<<NTOK / 8, 256, 0, stream>>>(X, lnfg, lnfb, H, NTOK);

  // ---- phase decoder + gating ----
  gemm(stream, H, DMOD, ph1W, DMOD, ph1b, Q, DMOD, nullptr,
       NTOK, DMOD, DMOD, ACT_PRELU, pha);
  gemm(stream, Q, DMOD, ph2W, DPH, ph2b, PH, DPH, nullptr,
       NTOK, DPH, DMOD, ACT_NONE, nullptr);
  gemm(stream, PH, DPH, g1W, 64, g1b, Q, 64, nullptr,
       NTOK, 64, DPH, ACT_PRELU, g1a);
  gemm(stream, Q, 64, g2W, 64, g2b, CTX, 64, nullptr,
       NTOK, 64, 64, ACT_PRELU, g2a);
  gemm(stream, CTX, 64, g3W, NEXP, g3b, GL, NEXP, nullptr,
       NTOK, NEXP, 64, ACT_NONE, nullptr);
  softmax8_kernel<<<NTOK / 256, 256, 0, stream>>>(GL, GW, NTOK);

  // ---- soft-MoE motion decoder ----
  moe(stream, H, moe0W, moe0b, GW, Q, NTOK, DMOD, DMOD, ACT_PRELU, moea1);
  moe(stream, Q, moe1W, moe1b, GW, CTX, NTOK, DMOD, DMOD, ACT_PRELU, moea2);
  moe(stream, CTX, moe2W, moe2b, GW, OUT, NTOK, MF, DMOD, ACT_NONE, nullptr);
}
